// MainNet_73177652789393
// MI455X (gfx1250) — matrix operand order repaired and checked by reading, not compiled, not run on hardware
//
#include <hip/hip_runtime.h>
#include <hip/hip_bf16.h>

typedef __attribute__((ext_vector_type(16))) _Float16 v16h;
typedef __attribute__((ext_vector_type(8)))  _Float16 v8h;
typedef __attribute__((ext_vector_type(8)))  float    v8f;

#define HH   64
#define WW   64
#define NPX  4096   // 64*64
#define NS   4
#define CNc  64
#define NSZ  24
#define DD   8

// padded (f16) weight strides
#define KPF  192    // flow   K=162
#define KPC  32     // ctx    K=27
#define KPP  64     // psv    K=54
#define KPV  4672   // vref   K=4635

static __device__ __forceinline__ int clampi(int v, int lo, int hi) {
    return v < lo ? lo : (v > hi ? hi : v);
}

// ---------------------------------------------------------------------------
// Inner MMA over one K-phase; everything is aligned wide loads.
//   aB = f16 weights row base  + k0 + g*16  (global, padded stride)
//   pB = panel + (ty*16+tx)*KPH + g*16      (LDS, fragment-ordered)
// A frag: k = t*32 + g*16 + e              -> two v8h global loads / step
// B frag: k = t*32 + g*16 + e              -> two v8h LDS loads / step
// ---------------------------------------------------------------------------
template <int KPH>
static __device__ __forceinline__ v8f panel_mma(const _Float16* __restrict__ aB,
                                                const _Float16* __restrict__ pB,
                                                v8f acc) {
#pragma unroll
    for (int t = 0; t < KPH / 32; ++t) {
        v8h alo = *(const v8h*)(aB + t * 32);
        v8h ahi = *(const v8h*)(aB + t * 32 + 8);
        v8h blo = *(const v8h*)(pB + t * 32);
        v8h bhi = *(const v8h*)(pB + t * 32 + 8);
        v16h A, B;
#pragma unroll
        for (int e = 0; e < 8; ++e) {
            A[e] = alo[e];  A[e + 8] = ahi[e];
            B[e] = blo[e];  B[e + 8] = bhi[e];
        }
        acc = __builtin_amdgcn_wmma_f32_16x16x32_f16(
            false, A, false, B, (short)0, acc, false, false);
    }
    return acc;
}

// ---------------------------------------------------------------------------
// im2col panel build (LDS halo -> LDS panel, fragment order).  thread=(ty,tx).
// halo layout [cin][18][18], zero-padded; panel [(ty*16+tx)*KPH + kk].
// Const-div by 9/3 -> mul+shift; staging only, amortized over 4 M-tiles.
// ---------------------------------------------------------------------------
template <int KPH>
static __device__ __forceinline__ void build_panel(_Float16* __restrict__ pan,
                                                   const _Float16* __restrict__ halo,
                                                   int tid, int k0) {
    const int ty = tid >> 4, tx = tid & 15;
    const _Float16* hB = halo + ty * 18 + tx;
    _Float16* pB = pan + tid * KPH;
#pragma unroll
    for (int kk = 0; kk < KPH; ++kk) {
        int k   = k0 + kk;
        int cin = k / 9;
        int tap = k - cin * 9;
        int dy  = tap / 3;
        int dx  = tap - dy * 3;
        pB[kk] = hB[cin * 324 + dy * 18 + dx];
    }
}

// ---------------------------------------------------------------------------
// Weight prep: convert all conv weights to f16, zero-padded to KP strides.
// ---------------------------------------------------------------------------
__global__ void k_prep_w(const float* __restrict__ fw, const float* __restrict__ cw,
                         const float* __restrict__ pw, const float* __restrict__ vw,
                         _Float16* __restrict__ wf, _Float16* __restrict__ wc,
                         _Float16* __restrict__ wp, _Float16* __restrict__ wv) {
    int i = blockIdx.x * 256 + threadIdx.x;
    if (i < 64 * KPF) {
        int m = i / KPF, k = i - m * KPF;
        wf[i] = (k < 162) ? (_Float16)fw[m * 162 + k] : (_Float16)0.0f;
        return;
    }
    i -= 64 * KPF;
    if (i < 64 * KPC) {
        int m = i >> 5, k = i & 31;
        wc[i] = (k < 27) ? (_Float16)cw[m * 27 + k] : (_Float16)0.0f;
        return;
    }
    i -= 64 * KPC;
    if (i < 64 * KPP) {
        int m = i >> 6, k = i & 63;
        wp[i] = (k < 54) ? (_Float16)pw[m * 54 + k] : (_Float16)0.0f;
        return;
    }
    i -= 64 * KPP;
    if (i < 16 * KPV) {
        int m = i / KPV, k = i - m * KPV;
        wv[i] = (m < 3 && k < 4635) ? (_Float16)vw[m * 4635 + k] : (_Float16)0.0f;
    }
}

// ---------------------------------------------------------------------------
// flowRefNet conv: concat[sf(6), sv(3), wv(9)], K=162, 3 phases of 64.
// Block = (img, 16x16 tile); 8 waves x 8 jobs; acc carried across phases.
// ---------------------------------------------------------------------------
__global__ void k_conv_flow(const float* __restrict__ sf,
                            const float* __restrict__ sv,
                            const float* __restrict__ wv,
                            const _Float16* __restrict__ w16,
                            const float* __restrict__ bias,
                            float* __restrict__ out) {
    __shared__ _Float16 halo[22 * 324];
    __shared__ _Float16 pan[256 * 64];
    const int tid = threadIdx.x;
    const int b   = blockIdx.x >> 4;
    const int tl  = blockIdx.x & 15;
    const int Y0  = (tl >> 2) << 4, X0 = (tl & 3) << 4;

    for (int i = tid; i < 22 * 324; i += 256) {
        int c = i / 324, r = i - c * 324;
        int hy = r / 18, hx = r - hy * 18;
        int py = Y0 + hy - 1, px = X0 + hx - 1;
        float v = 0.0f;
        if (c < 18 && py >= 0 && py < HH && px >= 0 && px < WW) {
            int pix = py * WW + px;
            v = (c < 6) ? sf[(b * 6 + c) * NPX + pix]
              : (c < 9) ? sv[(b * 3 + (c - 6)) * NPX + pix]
                        : wv[(b * 9 + (c - 9)) * NPX + pix];
        }
        halo[i] = (_Float16)v;
    }

    const int wid = tid >> 5, lane = tid & 31, l15 = lane & 15, g = lane >> 4;
    v8f acc[8];
#pragma unroll
    for (int a = 0; a < 8; ++a) acc[a] = (v8f){};

    for (int ph = 0; ph < 3; ++ph) {
        __syncthreads();
        build_panel<64>(pan, halo, tid, ph * 64);
        __syncthreads();
#pragma unroll
        for (int a = 0; a < 8; ++a) {
            int j = wid + a * 8;
            int mt = j >> 4, ty = j & 15;
            acc[a] = panel_mma<64>(w16 + (mt * 16 + l15) * KPF + ph * 64 + g * 16,
                                   pan + (ty * 16 + l15) * 64 + g * 16, acc[a]);
        }
    }
#pragma unroll
    for (int a = 0; a < 8; ++a) {
        int j = wid + a * 8;
        int mt = j >> 4, ty = j & 15;
        int pixo = (Y0 + ty) * WW + X0 + l15;
#pragma unroll
        for (int jj = 0; jj < 8; ++jj) {
            int oc = mt * 16 + jj + 8 * g;
            float v = acc[a][jj] + bias[oc];
            out[(b * CNc + oc) * NPX + pixo] = v > 0.0f ? v : 0.0f;
        }
    }
}

// ---------------------------------------------------------------------------
// ctxNet conv: sv(3), K=27, single phase of 32.
// ---------------------------------------------------------------------------
__global__ void k_conv_ctx(const float* __restrict__ sv,
                           const _Float16* __restrict__ w16,
                           const float* __restrict__ bias,
                           float* __restrict__ out) {
    __shared__ _Float16 halo[4 * 324];
    __shared__ _Float16 pan[256 * 32];
    const int tid = threadIdx.x;
    const int b   = blockIdx.x >> 4;
    const int tl  = blockIdx.x & 15;
    const int Y0  = (tl >> 2) << 4, X0 = (tl & 3) << 4;

    for (int i = tid; i < 4 * 324; i += 256) {
        int c = i / 324, r = i - c * 324;
        int hy = r / 18, hx = r - hy * 18;
        int py = Y0 + hy - 1, px = X0 + hx - 1;
        float v = 0.0f;
        if (c < 3 && py >= 0 && py < HH && px >= 0 && px < WW)
            v = sv[(b * 3 + c) * NPX + py * WW + px];
        halo[i] = (_Float16)v;
    }
    __syncthreads();
    build_panel<32>(pan, halo, tid, 0);
    __syncthreads();

    const int wid = tid >> 5, lane = tid & 31, l15 = lane & 15, g = lane >> 4;
#pragma unroll
    for (int a = 0; a < 8; ++a) {
        int j = wid + a * 8;
        int mt = j >> 4, ty = j & 15;
        v8f acc = {};
        acc = panel_mma<32>(w16 + (mt * 16 + l15) * KPC + g * 16,
                            pan + (ty * 16 + l15) * 32 + g * 16, acc);
        int pixo = (Y0 + ty) * WW + X0 + l15;
#pragma unroll
        for (int jj = 0; jj < 8; ++jj) {
            int oc = mt * 16 + jj + 8 * g;
            float v = acc[jj] + bias[oc];
            out[(b * CNc + oc) * NPX + pixo] = v > 0.0f ? v : 0.0f;
        }
    }
}

// ---------------------------------------------------------------------------
// Epipolar gather + weightNet softmax + interps (VALU by nature:
// data-dependent indices).  One thread per (source s, pixel p).
// ---------------------------------------------------------------------------
__global__ void k_epi(const float* __restrict__ ce, const float* __restrict__ ctx,
                      const float* __restrict__ sv, const float* __restrict__ sf,
                      const int*   __restrict__ nxy,
                      const float* __restrict__ spa, const float* __restrict__ ang,
                      const float* __restrict__ wnW, const float* __restrict__ wnB,
                      const float* __restrict__ cnW,
                      float* __restrict__ out_weight,   // [ns,np,24]
                      float* __restrict__ out_iv,       // [ns,3,np]
                      float* __restrict__ ws_iv,        // [ns,np,3]
                      float* __restrict__ ws_ictx,      // [ns*64,np]
                      float* __restrict__ ws_clog) {    // [ns,np]
    int gid = blockIdx.x * blockDim.x + threadIdx.x;
    if (gid >= NS * NPX) return;
    int s = gid >> 12, p = gid & (NPX - 1);

    int idxn[NSZ];
#pragma unroll
    for (int e = 0; e < 8; ++e) {
        int xx = nxy[((s * NPX + p) * 8 + e) * 2 + 0];
        int yy = nxy[((s * NPX + p) * 8 + e) * 2 + 1];
        xx = clampi(xx, 0, WW - 1);
#pragma unroll
        for (int dyi = 0; dyi < 3; ++dyi)
            idxn[e * 3 + dyi] = clampi(yy + dyi - 1, 0, HH - 1) * WW + xx;
    }

    float wce[NSZ], w2ce[NSZ];
#pragma unroll
    for (int n = 0; n < NSZ; ++n) { wce[n] = 0.0f; w2ce[n] = 0.0f; }
    float mudW = 0.0f, vadW = 0.0f, mudC = 0.0f, vadC = 0.0f;
    for (int c = 0; c < CNc; ++c) {
        const float* base = ce + (s * CNc + c) * NPX;
        float wn = wnW[6 + c], cn = cnW[6 + c];
        float ssum = 0.0f, ssq = 0.0f;
#pragma unroll
        for (int n = 0; n < NSZ; ++n) {
            float v = base[idxn[n]];
            ssum += v;  ssq += v * v;
            wce[n]  += v * wn;
            w2ce[n] += v * cn;
        }
        float mu  = ssum * (1.0f / 24.0f);
        float var = (ssq - ssum * mu) * (1.0f / 23.0f);
        mudW += mu * wnW[70 + c];  vadW += var * wnW[134 + c];
        mudC += mu * cnW[70 + c];  vadC += var * cnW[134 + c];
    }

    float logit[NSZ], clog[NSZ], lmax = -1e30f;
    for (int n = 0; n < NSZ; ++n) {
        float lw = wnB[0] + wce[n] + mudW + vadW;
        float lc = w2ce[n] + mudC + vadC;
#pragma unroll
        for (int j = 0; j < 6; ++j) {
            float v = sf[(s * 6 + j) * NPX + idxn[n]];
            lw += v * wnW[j];  lc += v * cnW[j];
        }
#pragma unroll
        for (int j = 0; j < 2; ++j) {
            float v = spa[((s * 2 + j) * NSZ + n) * NPX + p];
            lw += v * wnW[198 + j];  lc += v * cnW[198 + j];
        }
#pragma unroll
        for (int j = 0; j < 6; ++j) {
            float v = ang[((s * 6 + j) * NSZ + n) * NPX + p];
            lw += v * wnW[200 + j];  lc += v * cnW[200 + j];
        }
        logit[n] = lw;  clog[n] = lc;
        lmax = fmaxf(lmax, lw);
    }

    float wsm[NSZ], sum = 0.0f;
    for (int n = 0; n < NSZ; ++n) { wsm[n] = expf(logit[n] - lmax); sum += wsm[n]; }
    float inv = 1.0f / sum, celog = 0.0f;
    for (int n = 0; n < NSZ; ++n) {
        float wv = wsm[n] * inv;
        wsm[n] = wv;
        out_weight[(s * NPX + p) * NSZ + n] = wv;
        celog += wv * clog[n];
    }
    ws_clog[s * NPX + p] = celog;

#pragma unroll
    for (int c = 0; c < 3; ++c) {
        const float* base = sv + (s * 3 + c) * NPX;
        float a = 0.0f;
#pragma unroll
        for (int n = 0; n < NSZ; ++n) a += base[idxn[n]] * wsm[n];
        ws_iv[(s * NPX + p) * 3 + c] = a;
        out_iv[(s * 3 + c) * NPX + p] = a;
    }
    for (int c = 0; c < CNc; ++c) {
        const float* base = ctx + (s * CNc + c) * NPX;
        float a = 0.0f;
#pragma unroll
        for (int n = 0; n < NSZ; ++n) a += base[idxn[n]] * wsm[n];
        ws_ictx[(s * CNc + c) * NPX + p] = a;
    }
}

// ---------------------------------------------------------------------------
// confNet softmax over sources + blend (cnet bias cancels in the softmax).
// ---------------------------------------------------------------------------
__global__ void k_blend(const float* __restrict__ ws_clog,
                        const float* __restrict__ ws_iv,
                        float* __restrict__ out_conf,   // [ns,np]
                        float* __restrict__ out_blend,  // [3,np]
                        float* __restrict__ ws_bl) {    // [3,np]
    int p = blockIdx.x * blockDim.x + threadIdx.x;
    if (p >= NPX) return;
    float cl[NS], m = -1e30f;
#pragma unroll
    for (int s = 0; s < NS; ++s) { cl[s] = ws_clog[s * NPX + p]; m = fmaxf(m, cl[s]); }
    float sum = 0.0f;
#pragma unroll
    for (int s = 0; s < NS; ++s) { cl[s] = expf(cl[s] - m); sum += cl[s]; }
    float inv = 1.0f / sum;
#pragma unroll
    for (int s = 0; s < NS; ++s) { cl[s] *= inv; out_conf[s * NPX + p] = cl[s]; }
#pragma unroll
    for (int c = 0; c < 3; ++c) {
        float a = 0.0f;
#pragma unroll
        for (int s = 0; s < NS; ++s) a += ws_iv[(s * NPX + p) * 3 + c] * cl[s];
        out_blend[c * NPX + p] = a;
        ws_bl[c * NPX + p] = a;
    }
}

// ---------------------------------------------------------------------------
// psvNet: halo ch [0..2]=psv slice d (restaged per d), [3..5]=blended,
// [6..7]=zero.  K=54 -> one phase of 64.  Mean over D in registers.
// ---------------------------------------------------------------------------
__global__ void k_conv_psv(const float* __restrict__ psvs,
                           const float* __restrict__ blended,
                           const _Float16* __restrict__ w16,
                           const float* __restrict__ bias,
                           float* __restrict__ psv_feat) {   // [ns*64,np]
    __shared__ _Float16 halo[8 * 324];
    __shared__ _Float16 pan[256 * 64];
    const int tid = threadIdx.x;
    const int s   = blockIdx.x >> 4;
    const int tl  = blockIdx.x & 15;
    const int Y0  = (tl >> 2) << 4, X0 = (tl & 3) << 4;

    for (int i = tid; i < 5 * 324; i += 256) {       // channels 3..7 once
        int c = 3 + i / 324, r = i - (c - 3) * 324;
        int hy = r / 18, hx = r - hy * 18;
        int py = Y0 + hy - 1, px = X0 + hx - 1;
        float v = 0.0f;
        if (c < 6 && py >= 0 && py < HH && px >= 0 && px < WW)
            v = blended[(c - 3) * NPX + py * WW + px];
        halo[c * 324 + r] = (_Float16)v;
    }

    const int wid = tid >> 5, lane = tid & 31, l15 = lane & 15, g = lane >> 4;
    float macc[8][8];
#pragma unroll
    for (int a = 0; a < 8; ++a)
#pragma unroll
        for (int jj = 0; jj < 8; ++jj) macc[a][jj] = 0.0f;

    for (int d = 0; d < DD; ++d) {
        __syncthreads();
        for (int i = tid; i < 3 * 324; i += 256) {   // channels 0..2 per d
            int c = i / 324, r = i - c * 324;
            int hy = r / 18, hx = r - hy * 18;
            int py = Y0 + hy - 1, px = X0 + hx - 1;
            float v = 0.0f;
            if (py >= 0 && py < HH && px >= 0 && px < WW)
                v = psvs[((s * 3 + c) * DD + d) * NPX + py * WW + px];
            halo[i] = (_Float16)v;
        }
        __syncthreads();
        build_panel<64>(pan, halo, tid, 0);
        __syncthreads();
#pragma unroll
        for (int a = 0; a < 8; ++a) {
            int j = wid + a * 8;
            int mt = j >> 4, ty = j & 15;
            v8f acc = {};
            acc = panel_mma<64>(w16 + (mt * 16 + l15) * KPP + g * 16,
                                pan + (ty * 16 + l15) * 64 + g * 16, acc);
#pragma unroll
            for (int jj = 0; jj < 8; ++jj) {
                float v = acc[jj] + bias[mt * 16 + jj + 8 * g];
                macc[a][jj] += (v > 0.0f ? v : 0.0f) * 0.125f;
            }
        }
    }
#pragma unroll
    for (int a = 0; a < 8; ++a) {
        int j = wid + a * 8;
        int mt = j >> 4, ty = j & 15;
        int pixo = (Y0 + ty) * WW + X0 + l15;
#pragma unroll
        for (int jj = 0; jj < 8; ++jj) {
            int oc = mt * 16 + jj + 8 * g;
            psv_feat[(s * CNc + oc) * NPX + pixo] = macc[a][jj];
        }
    }
}

// ---------------------------------------------------------------------------
// vref conv: 515 ch chunked by 32 (halo 20.7KB); 9 phases of 32 per chunk.
// Acc carried across all chunks/phases.  3 out channels (rows 3..15 zero).
// ---------------------------------------------------------------------------
__global__ void k_conv_vref(const float* __restrict__ blended,
                            const float* __restrict__ ictx,
                            const float* __restrict__ psvf,
                            const _Float16* __restrict__ w16,
                            const float* __restrict__ bias,
                            float* __restrict__ out0) {      // [3,np]
    __shared__ _Float16 halo[32 * 324];
    __shared__ _Float16 pan[256 * 32];
    const int tid = threadIdx.x;
    const int tl  = blockIdx.x;
    const int Y0  = (tl >> 2) << 4, X0 = (tl & 3) << 4;
    const int wid = tid >> 5, lane = tid & 31, l15 = lane & 15, g = lane >> 4;

    v8f acc0 = {}, acc1 = {};
    for (int c0 = 0; c0 < 515; c0 += 32) {
        __syncthreads();
        for (int i = tid; i < 32 * 324; i += 256) {
            int c = i / 324, r = i - c * 324;
            int hy = r / 18, hx = r - hy * 18;
            int py = Y0 + hy - 1, px = X0 + hx - 1;
            int cg = c0 + c;
            float v = 0.0f;
            if (cg < 515 && py >= 0 && py < HH && px >= 0 && px < WW) {
                int pix = py * WW + px;
                v = (cg < 3)   ? blended[cg * NPX + pix]
                  : (cg < 259) ? ictx[(cg - 3) * NPX + pix]
                               : psvf[(cg - 259) * NPX + pix];
            }
            halo[i] = (_Float16)v;
        }
        for (int ph = 0; ph < 9; ++ph) {
            __syncthreads();
            build_panel<32>(pan, halo, tid, ph * 32);
            __syncthreads();
            const _Float16* aB = w16 + l15 * KPV + c0 * 9 + ph * 32 + g * 16;
            acc0 = panel_mma<32>(aB, pan + (wid * 16 + l15) * 32 + g * 16, acc0);
            acc1 = panel_mma<32>(aB, pan + ((wid + 8) * 16 + l15) * 32 + g * 16, acc1);
        }
    }

#pragma unroll
    for (int jj = 0; jj < 8; ++jj) {
        int oc = jj + 8 * g;
        if (oc < 3) {
            out0[oc * NPX + (Y0 + wid) * WW + X0 + l15]     = acc0[jj] + bias[oc];
            out0[oc * NPX + (Y0 + wid + 8) * WW + X0 + l15] = acc1[jj] + bias[oc];
        }
    }
}

// ---------------------------------------------------------------------------
extern "C" void kernel_launch(void* const* d_in, const int* in_sizes, int n_in,
                              void* d_out, int out_size, void* d_ws, size_t ws_size,
                              hipStream_t stream) {
    (void)in_sizes; (void)n_in; (void)out_size; (void)ws_size;
    const float* sv     = (const float*)d_in[0];
    const float* sf     = (const float*)d_in[1];
    const float* psvs   = (const float*)d_in[2];
    const float* wv     = (const float*)d_in[3];
    const int*   nxy    = (const int*)  d_in[5];
    const float* spa    = (const float*)d_in[6];
    const float* ang    = (const float*)d_in[7];
    const float* flow_w = (const float*)d_in[9];
    const float* flow_b = (const float*)d_in[10];
    const float* ctx_w  = (const float*)d_in[11];
    const float* ctx_b  = (const float*)d_in[12];
    const float* wnet_w = (const float*)d_in[13];
    const float* wnet_b = (const float*)d_in[14];
    const float* cnet_w = (const float*)d_in[15];
    const float* psv_w  = (const float*)d_in[17];
    const float* psv_b  = (const float*)d_in[18];
    const float* vref_w = (const float*)d_in[19];
    const float* vref_b = (const float*)d_in[20];

    float* out     = (float*)d_out;
    float* o_ref   = out;                 // [1,3,64,64]       12288
    float* o_blend = out + 12288;         // [1,3,64,64]       12288
    float* o_iv    = out + 24576;         // [1,4,3,64,64]     49152
    float* o_wgt   = out + 73728;         // [1,4,64,64,24]    393216
    float* o_conf  = out + 466944;        // [1,4,64,64]       16384

    float* ws      = (float*)d_ws;
    float* ws_ce   = ws;                          // 4*64*4096
    float* ws_ctx  = ws_ce   + NS * CNc * NPX;    // 4*64*4096
    float* ws_ictx = ws_ctx  + NS * CNc * NPX;    // 4*64*4096
    float* ws_psvf = ws_ictx + NS * CNc * NPX;    // 4*64*4096
    float* ws_iv   = ws_psvf + NS * CNc * NPX;    // 4*4096*3
    float* ws_clog = ws_iv   + NS * NPX * 3;      // 4*4096
    float* ws_bl   = ws_clog + NS * NPX;          // 3*4096
    _Float16* wf16 = (_Float16*)(ws_bl + 3 * NPX);   // 64*192 halves, 16B aligned
    _Float16* wc16 = wf16 + 64 * KPF;                // 64*32
    _Float16* wp16 = wc16 + 64 * KPC;                // 64*64
    _Float16* wv16 = wp16 + 64 * KPP;                // 16*4672

    const int prepN = 64 * KPF + 64 * KPC + 64 * KPP + 16 * KPV;
    k_prep_w   <<<(prepN + 255) / 256, 256, 0, stream>>>(flow_w, ctx_w, psv_w, vref_w,
                                                         wf16, wc16, wp16, wv16);
    k_conv_flow<<<64, 256, 0, stream>>>(sf, sv, wv, wf16, flow_b, ws_ce);
    k_conv_ctx <<<64, 256, 0, stream>>>(sv, wc16, ctx_b, ws_ctx);
    k_epi      <<<64, 256, 0, stream>>>(ws_ce, ws_ctx, sv, sf, nxy, spa, ang,
                                        wnet_w, wnet_b, cnet_w,
                                        o_wgt, o_iv, ws_iv, ws_ictx, ws_clog);
    k_blend    <<<16, 256, 0, stream>>>(ws_clog, ws_iv, o_conf, o_blend, ws_bl);
    k_conv_psv <<<64, 256, 0, stream>>>(psvs, ws_bl, wp16, psv_b, ws_psvf);
    k_conv_vref<<<16, 256, 0, stream>>>(ws_bl, ws_ictx, ws_psvf, wv16, vref_b, o_ref);
}
